// Attention_46231027974215
// MI455X (gfx1250) — compile-verified, matrix operand order checked by reading, not run on hardware
//
#include <hip/hip_runtime.h>
#include <hip/hip_bf16.h>
#include <stdint.h>

typedef __attribute__((ext_vector_type(16))) __bf16 v16bf;
typedef __attribute__((ext_vector_type(2)))  __bf16 v2bf;
typedef __attribute__((ext_vector_type(8)))  float  v8f;

#define B_SZ    4
#define SEQ     4096
#define DHEAD   64
#define QTILE   64          // query rows per workgroup
#define KTILE   64          // keys per iteration
#define NWAVES  4
#define LSTRIDE 72          // bf16 elements per LDS row (pad vs. bank conflicts)

// ---- bf16 conversion: HW packed cvt if present, else v_perm_b32 packing ----
__device__ __forceinline__ unsigned int f2bf_pk(float lo, float hi) {
#if __has_builtin(__builtin_amdgcn_cvt_pk_bf16_f32)
    union { v2bf v; unsigned int u; } c;
    c.v = __builtin_amdgcn_cvt_pk_bf16_f32(lo, hi);
    return c.u;
#else
    union { float f; unsigned int u; } a, b;
    a.f = lo; b.f = hi;
    // D = { hi16(b), hi16(a) } : one v_perm_b32 packs two bf16 (truncate)
    return __builtin_amdgcn_perm(b.u, a.u, 0x07060302u);
#endif
}

__device__ __forceinline__ unsigned short f2bf1(float f) {
#if __has_builtin(__builtin_amdgcn_cvt_pk_bf16_f32)
    return (unsigned short)(f2bf_pk(f, f) & 0xFFFFu);
#else
    union { float f; unsigned int u; } c; c.f = f;
    return (unsigned short)(c.u >> 16);   // truncating bf16, single shift
#endif
}

// ---- fused DPP butterfly ops: one VOP2-DPP instruction per step ----
// masks {1,2,7,15} generate the 16-lane group (stays inside each half-wave)
__device__ __forceinline__ float max_dpp_x1(float x) {
    float r;
    asm("v_max_num_f32_dpp %0, %1, %2 quad_perm:[1,0,3,2] row_mask:0xf bank_mask:0xf bound_ctrl:1"
        : "=v"(r) : "v"(x), "v"(x));
    return r;
}
__device__ __forceinline__ float max_dpp_x2(float x) {
    float r;
    asm("v_max_num_f32_dpp %0, %1, %2 quad_perm:[2,3,0,1] row_mask:0xf bank_mask:0xf bound_ctrl:1"
        : "=v"(r) : "v"(x), "v"(x));
    return r;
}
__device__ __forceinline__ float max_dpp_x7(float x) {
    float r;
    asm("v_max_num_f32_dpp %0, %1, %2 row_half_mirror row_mask:0xf bank_mask:0xf bound_ctrl:1"
        : "=v"(r) : "v"(x), "v"(x));
    return r;
}
__device__ __forceinline__ float max_dpp_x15(float x) {
    float r;
    asm("v_max_num_f32_dpp %0, %1, %2 row_mirror row_mask:0xf bank_mask:0xf bound_ctrl:1"
        : "=v"(r) : "v"(x), "v"(x));
    return r;
}
__device__ __forceinline__ float add_dpp_x1(float x) {
    float r;
    asm("v_add_f32_dpp %0, %1, %2 quad_perm:[1,0,3,2] row_mask:0xf bank_mask:0xf bound_ctrl:1"
        : "=v"(r) : "v"(x), "v"(x));
    return r;
}
__device__ __forceinline__ float add_dpp_x2(float x) {
    float r;
    asm("v_add_f32_dpp %0, %1, %2 quad_perm:[2,3,0,1] row_mask:0xf bank_mask:0xf bound_ctrl:1"
        : "=v"(r) : "v"(x), "v"(x));
    return r;
}
__device__ __forceinline__ float add_dpp_x7(float x) {
    float r;
    asm("v_add_f32_dpp %0, %1, %2 row_half_mirror row_mask:0xf bank_mask:0xf bound_ctrl:1"
        : "=v"(r) : "v"(x), "v"(x));
    return r;
}
__device__ __forceinline__ float add_dpp_x15(float x) {
    float r;
    asm("v_add_f32_dpp %0, %1, %2 row_mirror row_mask:0xf bank_mask:0xf bound_ctrl:1"
        : "=v"(r) : "v"(x), "v"(x));
    return r;
}
__device__ __forceinline__ float half_reduce_max(float x) {
    x = max_dpp_x1(x); x = max_dpp_x2(x); x = max_dpp_x7(x); x = max_dpp_x15(x);
    return x;
}
__device__ __forceinline__ float half_reduce_sum(float x) {
    x = add_dpp_x1(x); x = add_dpp_x2(x); x = add_dpp_x7(x); x = add_dpp_x15(x);
    return x;
}

union Frag16 { v16bf v; unsigned short h[16]; unsigned int u[8]; };

// ---- one 64-key flash-attention tile; DIAG selects the masked diagonal path ----
template<bool DIAG>
__device__ __forceinline__ void attn_tile(
    const int kBase, const bool doPrefetch,
    const float* __restrict__ Kb, const float* __restrict__ Vb,
    const unsigned char* __restrict__ padb,
    unsigned short* sK, unsigned short* sVT, unsigned short* sP, float* sPad,
    const Frag16 (&aQ)[2], v8f (&acc)[4], float (&mstat)[8], float (&lstat)[8],
    const int tid, const int wave, const int half, const int l16,
    const int hb, const int qRow0)
{
    // ---- cooperative stage: K -> sK (bf16), V -> sVT (bf16, transposed) ----
    // each thread: 2 key rows x 16 dims; V pairs (2 keys, same dim) pack into
    // one b32 so the transposed store is aligned and shift-free.
    {
        const int r2 = (tid >> 2) * 2;       // first of two key rows (0,2,..,62)
        const int c0 = (tid & 3) * 16;       // 16-dim slab
        const float* kr0 = Kb + (size_t)(kBase + r2) * DHEAD + c0;
        const float* vr0 = Vb + (size_t)(kBase + r2) * DHEAD + c0;
        #pragma unroll
        for (int c = 0; c < 16; c += 4) {
            const float4 k0 = *(const float4*)(kr0 + c);
            const float4 k1 = *(const float4*)(kr0 + DHEAD + c);
            const float4 v0 = *(const float4*)(vr0 + c);
            const float4 v1 = *(const float4*)(vr0 + DHEAD + c);
            const int col = c0 + c;
            *(unsigned int*)&sK[(r2 + 0) * LSTRIDE + col + 0] = f2bf_pk(k0.x, k0.y);
            *(unsigned int*)&sK[(r2 + 0) * LSTRIDE + col + 2] = f2bf_pk(k0.z, k0.w);
            *(unsigned int*)&sK[(r2 + 1) * LSTRIDE + col + 0] = f2bf_pk(k1.x, k1.y);
            *(unsigned int*)&sK[(r2 + 1) * LSTRIDE + col + 2] = f2bf_pk(k1.z, k1.w);
            *(unsigned int*)&sVT[(col + 0) * LSTRIDE + r2] = f2bf_pk(v0.x, v1.x);
            *(unsigned int*)&sVT[(col + 1) * LSTRIDE + r2] = f2bf_pk(v0.y, v1.y);
            *(unsigned int*)&sVT[(col + 2) * LSTRIDE + r2] = f2bf_pk(v0.z, v1.z);
            *(unsigned int*)&sVT[(col + 3) * LSTRIDE + r2] = f2bf_pk(v0.w, v1.w);
        }
        if (tid < KTILE)
            sPad[tid] = padb[kBase + tid] ? -1.0e30f : 0.0f;
    }
    // prefetch next tile into L2 while we compute (global_prefetch_b8)
    if (doPrefetch) {
        const int row = tid >> 1;
        __builtin_prefetch(Kb + (size_t)(kBase + KTILE + row) * DHEAD, 0, 1);
        __builtin_prefetch(Vb + (size_t)(kBase + KTILE + row) * DHEAD, 0, 1);
    }
    __syncthreads();

    // ---- S = Q * K^T : 4 key-blocks x 2 K-chunks, chained WMMA ----
    v8f s[4];
    #pragma unroll
    for (int nb = 0; nb < 4; ++nb) {
        Frag16 bK0, bK1;
        const int key = nb * 16 + l16;       // B layout: lane holds column n
        #pragma unroll
        for (int v = 0; v < 8; ++v) {
            const int k = (v >> 2) * 16 + hb + (v & 3) * 2;
            bK0.u[v] = *(const unsigned int*)&sK[key * LSTRIDE + k];
            bK1.u[v] = *(const unsigned int*)&sK[key * LSTRIDE + k + 32];
        }
        v8f c = {};
        c = __builtin_amdgcn_wmma_f32_16x16x32_bf16(false, aQ[0].v, false, bK0.v,
                                                    (short)0, c, false, false);
        c = __builtin_amdgcn_wmma_f32_16x16x32_bf16(false, aQ[1].v, false, bK1.v,
                                                    (short)0, c, false, false);
        s[nb] = c;
    }

    // ---- masks: padding always; causal compares only on the diagonal tile ----
    #pragma unroll
    for (int nb = 0; nb < 4; ++nb) {
        const float padAdj = sPad[nb * 16 + l16];
        if (DIAG) {
            const int key = kBase + nb * 16 + l16;
            #pragma unroll
            for (int v = 0; v < 8; ++v) {
                const int qr = qRow0 + v + 8 * half;
                float x = s[nb][v] + padAdj;
                if (key > qr) x = -1.0e30f;      // strict upper triangle
                s[nb][v] = x;
            }
        } else {
            #pragma unroll
            for (int v = 0; v < 8; ++v)
                s[nb][v] += padAdj;
        }
    }

    const float cexp = 0.18033688011112042f;     // log2(e) / sqrt(64)

    // ---- online softmax: row max over 64 keys (fused DPP butterfly) ----
    float mnew[8], nmb[8];
    #pragma unroll
    for (int v = 0; v < 8; ++v) {
        float x = fmaxf(fmaxf(s[0][v], s[1][v]), fmaxf(s[2][v], s[3][v]));
        mnew[v] = fmaxf(mstat[v], half_reduce_max(x));
        nmb[v]  = mnew[v] * (-cexp);             // -m*c for fma-folded exp arg
    }

    float rsum[8];
    #pragma unroll
    for (int v = 0; v < 8; ++v) rsum[v] = 0.0f;
    #pragma unroll
    for (int nb = 0; nb < 4; ++nb) {
        #pragma unroll
        for (int v = 0; v < 8; ++v) {
            const float p = __builtin_amdgcn_exp2f(__builtin_fmaf(s[nb][v], cexp, nmb[v]));
            s[nb][v] = p;
            rsum[v] += p;
        }
    }
    #pragma unroll
    for (int v = 0; v < 8; ++v) {
        const float x = half_reduce_sum(rsum[v]);
        const float fac = __builtin_amdgcn_exp2f(__builtin_fmaf(mstat[v], cexp, nmb[v]));
        lstat[v] = lstat[v] * fac + x;
        mstat[v] = mnew[v];
        acc[0][v] *= fac; acc[1][v] *= fac; acc[2][v] *= fac; acc[3][v] *= fac;
    }

    // ---- P: C layout -> LDS -> A layout (per-wave region, in-order DS) ----
    unsigned short* Pw = sP + wave * 16 * LSTRIDE;
    #pragma unroll
    for (int nb = 0; nb < 4; ++nb)
        #pragma unroll
        for (int v = 0; v < 8; ++v)
            Pw[(v + 8 * half) * LSTRIDE + nb * 16 + l16] = f2bf1(s[nb][v]);

    // ---- O += P * V : 2 K-chunks x 4 dim-blocks ----
    #pragma unroll
    for (int f = 0; f < 2; ++f) {
        Frag16 aP;
        #pragma unroll
        for (int v = 0; v < 8; ++v) {
            const int k = (v >> 2) * 16 + hb + (v & 3) * 2 + f * 32;
            aP.u[v] = *(const unsigned int*)&Pw[l16 * LSTRIDE + k];
        }
        #pragma unroll
        for (int nb = 0; nb < 4; ++nb) {
            Frag16 bV;
            #pragma unroll
            for (int v = 0; v < 8; ++v) {
                const int k = (v >> 2) * 16 + hb + (v & 3) * 2 + f * 32;
                bV.u[v] = *(const unsigned int*)&sVT[(nb * 16 + l16) * LSTRIDE + k];
            }
            acc[nb] = __builtin_amdgcn_wmma_f32_16x16x32_bf16(
                false, aP.v, false, bV.v, (short)0, acc[nb], false, false);
        }
    }
    __syncthreads();   // before next tile overwrites sK/sVT
}

__global__ __launch_bounds__(128, 1) void attn_fa_kernel(
    const float* __restrict__ Q, const float* __restrict__ K,
    const float* __restrict__ V, const unsigned char* __restrict__ pad,
    float* __restrict__ O)
{
    __shared__ __align__(16) unsigned short sK [KTILE  * LSTRIDE];      // K[key][dim]
    __shared__ __align__(16) unsigned short sVT[DHEAD  * LSTRIDE];      // V^T[dim][key]
    __shared__ __align__(16) unsigned short sP [NWAVES * 16 * LSTRIDE]; // per-wave P
    __shared__ float sPad[KTILE];

    const int qTile = blockIdx.x;         // 0..63
    const int b     = blockIdx.y;         // 0..3
    const int tid   = threadIdx.x;        // 0..127
    const int wave  = tid >> 5;
    const int lane  = tid & 31;
    const int half  = lane >> 4;          // which 16-lane half
    const int l16   = lane & 15;
    const int hb    = half * 8;           // K-offset base for this half (A/B layout)

    const int qRow0 = qTile * QTILE + wave * 16;   // first of this wave's 16 rows

    const size_t bOff = (size_t)b * SEQ * DHEAD;
    const float* Qb = Q + bOff;
    const float* Kb = K + bOff;
    const float* Vb = V + bOff;
    const unsigned char* padb = pad + (size_t)b * SEQ;

    // ---- Q A-fragments: 16x32 bf16, two chunks covering D=64 ----
    // A layout: lane m = l16 holds row m; VGPR v holds K = (v/4)*16 + hb + (v%4)*2 (+1)
    Frag16 aQ[2];
    #pragma unroll
    for (int f = 0; f < 2; ++f) {
        #pragma unroll
        for (int v = 0; v < 8; ++v) {
            const int k = (v >> 2) * 16 + hb + (v & 3) * 2 + f * 32;
            const float2 q2 = *(const float2*)(Qb + (size_t)(qRow0 + l16) * DHEAD + k);
            aQ[f].u[v] = f2bf_pk(q2.x, q2.y);
        }
    }

    // ---- persistent state: O accumulators + online-softmax row stats ----
    v8f acc[4] = {v8f{}, v8f{}, v8f{}, v8f{}};   // 16x64 fp32 output tile
    float mstat[8], lstat[8];                    // row r = v + 8*half
    #pragma unroll
    for (int v = 0; v < 8; ++v) { mstat[v] = -3.0e38f; lstat[v] = 0.0f; }

    const int jMax = qTile;               // causal: key tile index <= q tile index

    for (int j = 0; j < jMax; ++j)        // mask-free full tiles
        attn_tile<false>(j * KTILE, /*prefetch=*/true, Kb, Vb, padb,
                         sK, sVT, sP, sPad, aQ, acc, mstat, lstat,
                         tid, wave, half, l16, hb, qRow0);
    // diagonal tile with causal mask
    attn_tile<true>(jMax * KTILE, /*prefetch=*/false, Kb, Vb, padb,
                    sK, sVT, sP, sPad, aQ, acc, mstat, lstat,
                    tid, wave, half, l16, hb, qRow0);

    // ---- normalize and store O ----
    float invl[8];
    #pragma unroll
    for (int v = 0; v < 8; ++v) invl[v] = 1.0f / lstat[v];
    #pragma unroll
    for (int nb = 0; nb < 4; ++nb) {
        #pragma unroll
        for (int v = 0; v < 8; ++v) {
            const int qr = qRow0 + v + 8 * half;
            O[bOff + (size_t)qr * DHEAD + nb * 16 + l16] = acc[nb][v] * invl[v];
        }
    }
}

extern "C" void kernel_launch(void* const* d_in, const int* in_sizes, int n_in,
                              void* d_out, int out_size, void* d_ws, size_t ws_size,
                              hipStream_t stream) {
    const float* q = (const float*)d_in[0];
    const float* k = (const float*)d_in[1];
    const float* v = (const float*)d_in[2];
    const unsigned char* pad = (const unsigned char*)d_in[3];
    float* out = (float*)d_out;

    dim3 grid(SEQ / QTILE, B_SZ);
    dim3 block(NWAVES * 32);
    attn_fa_kernel<<<grid, block, 0, stream>>>(q, k, v, pad, out);
}
